// MyModel_87222195848138
// MI455X (gfx1250) — compile-verified, matrix operand order checked by reading, not run on hardware
//
#include <hip/hip_runtime.h>

// MI455X / gfx1250, wave32. bf16 WMMA for the two big GEMMs, async
// global->LDS staging (ASYNCcnt) with LDS double buffering in GEMM1.
// Algebraic reorderings: (A_norm @ x) @ Wmu^T == A_norm @ (x @ Wmu^T), and
// (A+I)@h0' == A@h0' + h0' (identity folded into the epilogue).

typedef __attribute__((ext_vector_type(16))) __bf16 v16bf;
typedef __attribute__((ext_vector_type(8)))  __bf16 v8bf;
typedef __attribute__((ext_vector_type(8)))  float  v8f;
typedef __attribute__((ext_vector_type(4)))  int    v4i;

#define NDIM  2048
#define BATCH 8
#define DIN   256

#if __has_builtin(__builtin_amdgcn_global_load_async_to_lds_b128)
#define HAVE_ASYNC_LDS 1
#else
#define HAVE_ASYNC_LDS 0
#endif

#if HAVE_ASYNC_LDS
#if __has_builtin(__builtin_amdgcn_s_wait_asynccnt)
#define WAIT_ASYNC(n) __builtin_amdgcn_s_wait_asynccnt(n)
#else
#define WAIT_ASYNC(n) asm volatile("s_wait_asynccnt %0" :: "n"(n) : "memory")
#endif
#else
#define WAIT_ASYNC(n) ((void)0)
#endif

// 16-byte global->LDS copy: async (no VGPR round trip) when available.
__device__ __forceinline__ void cp16(const void* g, void* l) {
#if HAVE_ASYNC_LDS
  v4i* gg = (v4i*)g;
  v4i* ll = (v4i*)l;
  __builtin_amdgcn_global_load_async_to_lds_b128(
      (__attribute__((address_space(1))) v4i*)gg,
      (__attribute__((address_space(3))) v4i*)ll, 0, 0);
#else
  *(uint4*)l = *(const uint4*)g;
#endif
}

__device__ __forceinline__ unsigned short f2bf(float f) {
  unsigned int u = __float_as_uint(f);
  unsigned int r = (u + 0x7FFFu + ((u >> 16) & 1u)) >> 16;  // RNE
  return (unsigned short)r;
}

// ---- d[b,i] = rsqrt(1 + sum_j A[b,i,j]); also emit Ab = bf16(A) ------------
__global__ void __launch_bounds__(256) deg_kernel(const float* __restrict__ A,
                                                  float* __restrict__ dvec,
                                                  unsigned short* __restrict__ Ab) {
  int wv = threadIdx.x >> 5, lane = threadIdx.x & 31;
  int rf = blockIdx.x * 8 + wv;                       // flat row in [0, B*N)
  const float4* row = (const float4*)(A + (size_t)rf * NDIM);
  unsigned short* orow = Ab + (size_t)rf * NDIM;
  float s = 0.f;
  #pragma unroll
  for (int c = 0; c < 16; ++c) {
    int q = c * 32 + lane;
    float4 v = row[q];
    s += v.x + v.y + v.z + v.w;
    unsigned int w0 = f2bf(v.x) | ((unsigned)f2bf(v.y) << 16);
    unsigned int w1 = f2bf(v.z) | ((unsigned)f2bf(v.w) << 16);
    *(uint2*)&orow[q * 4] = make_uint2(w0, w1);
  }
  #pragma unroll
  for (int m = 16; m; m >>= 1) s += __shfl_xor(s, m, 32);
  if (lane == 0) dvec[rf] = rsqrtf(s + 1.0f);
}

// ---- W_ext -> bf16 ---------------------------------------------------------
__global__ void __launch_bounds__(256) prep_w_kernel(const float* __restrict__ W,
                                                     unsigned short* __restrict__ Wb,
                                                     int n) {
  int i = blockIdx.x * blockDim.x + threadIdx.x;
  if (i < n) Wb[i] = f2bf(W[i]);
}

// ---- h0pT[b][f][j] = bf16(d[b,j] * h0[b,j,f]) ------------------------------
__global__ void __launch_bounds__(256) prep_h0_kernel(const float* __restrict__ h0,
                                                      const float* __restrict__ dvec,
                                                      unsigned short* __restrict__ h0pT) {
  __shared__ unsigned short tile[64 * DIN];           // 32 KB, [j_local][f]
  int b  = blockIdx.x >> 5;
  int j0 = (blockIdx.x & 31) * 64;
  int t  = threadIdx.x;
  int jr = t >> 2, q = t & 3;
  float dj = dvec[b * NDIM + j0 + jr];
  const float4* src = (const float4*)(h0 + ((size_t)b * NDIM + j0 + jr) * DIN + q * 64);
  #pragma unroll
  for (int c = 0; c < 16; ++c) {
    float4 v = src[c];
    unsigned int w0 = f2bf(v.x * dj) | ((unsigned)f2bf(v.y * dj) << 16);
    unsigned int w1 = f2bf(v.z * dj) | ((unsigned)f2bf(v.w * dj) << 16);
    *(uint2*)&tile[jr * DIN + q * 64 + c * 4] = make_uint2(w0, w1);
  }
  __syncthreads();
  size_t dst = ((size_t)b * DIN + t) * NDIM + j0;
  #pragma unroll
  for (int g = 0; g < 8; ++g) {
    unsigned int w[4];
    #pragma unroll
    for (int e = 0; e < 4; ++e) {
      unsigned short a = tile[(g * 8 + e * 2 + 0) * DIN + t];
      unsigned short c = tile[(g * 8 + e * 2 + 1) * DIN + t];
      w[e] = a | ((unsigned)c << 16);
    }
    *(uint4*)&h0pT[dst + g * 8] = make_uint4(w[0], w[1], w[2], w[3]);
  }
}

// ---- GEMM1: Hb[b,i,f] = bf16( d_i * ( (A@h0p)[i,f] + d_i*h0[b,i,f] ) ) -----
// 128x128 tile, K by 32, async double-buffered LDS staging, 8 waves of 32x64.
__global__ void __launch_bounds__(256) gemm1_kernel(const unsigned short* __restrict__ Ab,
                                                    const unsigned short* __restrict__ h0pT,
                                                    const float* __restrict__ h0,
                                                    const float* __restrict__ dvec,
                                                    unsigned short* __restrict__ Hb) {
  __shared__ unsigned short Atile[2][128 * 32];       // [m][k] row-major, 64B rows
  __shared__ unsigned short Btile[2][128 * 32];       // [n][k] row-major
  int b  = blockIdx.z;
  int m0 = blockIdx.y * 128;
  int n0 = blockIdx.x * 128;
  int t  = threadIdx.x;
  int wv = t >> 5, lane = t & 31;
  int wm = wv & 3, wn = wv >> 2;
  int l15 = lane & 15;
  int lo  = (lane < 16) ? 0 : 8;
  int hi8 = (lane >> 4) << 3;

  v8f acc[2][4];
  #pragma unroll
  for (int i = 0; i < 2; ++i)
    #pragma unroll
    for (int j = 0; j < 4; ++j)
      #pragma unroll
      for (int e = 0; e < 8; ++e) acc[i][j][e] = 0.f;

  const size_t Abase = (size_t)b * NDIM * NDIM;
  const size_t Bbase = (size_t)b * DIN * NDIM;
  int rq = t >> 2, q8 = (t & 3) * 8;                  // staging: row, 8-elem quarter

  // stage one K-slice (k0) into buffer buf: 4 x 16B per thread (2 A + 2 B)
  auto stage = [&](int k0, int buf) {
    #pragma unroll
    for (int it = 0; it < 2; ++it) {
      int row = rq + 64 * it;
      cp16(Ab + Abase + (size_t)(m0 + row) * NDIM + k0 + q8,
           &Atile[buf][row * 32 + q8]);
      cp16(h0pT + Bbase + (size_t)(n0 + row) * NDIM + k0 + q8,
           &Btile[buf][row * 32 + q8]);
    }
  };

  stage(0, 0);
  for (int kb = 0; kb < 64; ++kb) {
    int cur = kb & 1;
    if (kb + 1 < 64) {
      stage((kb + 1) * 32, cur ^ 1);
      WAIT_ASYNC(4);                                  // my stage(kb) done; kb+1 in flight
    } else {
      WAIT_ASYNC(0);
    }
    __syncthreads();                                  // stage(kb) visible from all waves

    v16bf afr[2], bfr[4];
    #pragma unroll
    for (int tm = 0; tm < 2; ++tm) {
      int r = wm * 32 + tm * 16 + l15;
      ((v8bf*)&afr[tm])[0] = *(const v8bf*)&Atile[cur][r * 32 + lo];
      ((v8bf*)&afr[tm])[1] = *(const v8bf*)&Atile[cur][r * 32 + lo + 16];
    }
    #pragma unroll
    for (int tn = 0; tn < 4; ++tn) {
      int cidx = wn * 64 + tn * 16 + l15;
      ((v8bf*)&bfr[tn])[0] = *(const v8bf*)&Btile[cur][cidx * 32 + lo];
      ((v8bf*)&bfr[tn])[1] = *(const v8bf*)&Btile[cur][cidx * 32 + lo + 16];
    }
    #pragma unroll
    for (int tm = 0; tm < 2; ++tm)
      #pragma unroll
      for (int tn = 0; tn < 4; ++tn)
        acc[tm][tn] = __builtin_amdgcn_wmma_f32_16x16x32_bf16(
            false, afr[tm], false, bfr[tn], (short)0, acc[tm][tn], false, false);
    if (kb + 1 < 64) __syncthreads();                 // all reads of cur done before reuse
  }

  // epilogue: +identity term (d_i*h0), scale by d_i, emit bf16 H
  #pragma unroll
  for (int tm = 0; tm < 2; ++tm) {
    #pragma unroll
    for (int e = 0; e < 8; ++e) {
      int gi = m0 + wm * 32 + tm * 16 + e + hi8;
      float ds = dvec[b * NDIM + gi];
      size_t rowo = ((size_t)b * NDIM + gi) * DIN;
      #pragma unroll
      for (int tn = 0; tn < 4; ++tn) {
        int gf = n0 + wn * 64 + tn * 16 + l15;
        float idterm = ds * h0[rowo + gf];            // h0p[i,f] = d_i*h0[i,f]
        Hb[rowo + gf] = f2bf(ds * (acc[tm][tn][e] + idterm));
      }
    }
  }
}

// ---- GEMM2 (fused): u[r] = sum_n relu(H[r]@Wext^T + bext)_n * Wmu_n --------
__global__ void __launch_bounds__(256) gemm2_kernel(const unsigned short* __restrict__ Hb,
                                                    const unsigned short* __restrict__ Wb,
                                                    const float* __restrict__ bext,
                                                    const float* __restrict__ Wmu,
                                                    float* __restrict__ u) {
  __shared__ unsigned short Arows[64 * DIN];          // 32 KB, [m][k] full K=256
  __shared__ unsigned short Btile[128 * 32];          // 8 KB, [n][k]
  __shared__ float bext_l[128];
  __shared__ float wmu_l[128];
  __shared__ float u_l[64];
  int t  = threadIdx.x;
  int r0 = blockIdx.x * 64;
  int wv = t >> 5, lane = t & 31;
  int wm = wv & 3, wn = wv >> 2;
  int l15 = lane & 15;
  int lo  = (lane < 16) ? 0 : 8;
  int hi8 = (lane >> 4) << 3;

  if (t < 64) u_l[t] = 0.f;
  {
    int row = t >> 2, q = (t & 3) * 64;
    #pragma unroll
    for (int c = 0; c < 8; ++c)
      cp16(Hb + (size_t)(r0 + row) * DIN + q + c * 8, &Arows[row * DIN + q + c * 8]);
    WAIT_ASYNC(0);
  }
  __syncthreads();

  float upart[8];
  #pragma unroll
  for (int e = 0; e < 8; ++e) upart[e] = 0.f;

  int srow = t >> 1, s16 = (t & 1) * 16;
  for (int nt = 0; nt < 16; ++nt) {
    int n0g = nt * 128;
    if (t < 128) { bext_l[t] = bext[n0g + t]; wmu_l[t] = Wmu[n0g + t]; }
    v8f acc[4];
    #pragma unroll
    for (int j = 0; j < 4; ++j)
      #pragma unroll
      for (int e = 0; e < 8; ++e) acc[j][e] = 0.f;

    for (int k0 = 0; k0 < DIN; k0 += 32) {
      cp16(Wb + (size_t)(n0g + srow) * DIN + k0 + s16, &Btile[srow * 32 + s16]);
      cp16(Wb + (size_t)(n0g + srow) * DIN + k0 + s16 + 8, &Btile[srow * 32 + s16 + 8]);
      WAIT_ASYNC(0);
      __syncthreads();

      v16bf afr, bfr[4];
      int r = wm * 16 + l15;
      ((v8bf*)&afr)[0] = *(const v8bf*)&Arows[r * DIN + k0 + lo];
      ((v8bf*)&afr)[1] = *(const v8bf*)&Arows[r * DIN + k0 + lo + 16];
      #pragma unroll
      for (int tn = 0; tn < 4; ++tn) {
        int cidx = wn * 64 + tn * 16 + l15;
        ((v8bf*)&bfr[tn])[0] = *(const v8bf*)&Btile[cidx * 32 + lo];
        ((v8bf*)&bfr[tn])[1] = *(const v8bf*)&Btile[cidx * 32 + lo + 16];
      }
      #pragma unroll
      for (int tn = 0; tn < 4; ++tn)
        acc[tn] = __builtin_amdgcn_wmma_f32_16x16x32_bf16(
            false, afr, false, bfr[tn], (short)0, acc[tn], false, false);
      __syncthreads();
    }
    // fused epilogue: relu(+bias), weighted reduction along N
    #pragma unroll
    for (int tn = 0; tn < 4; ++tn) {
      int cl = wn * 64 + tn * 16 + l15;
      float be = bext_l[cl], wmw = wmu_l[cl];
      #pragma unroll
      for (int e = 0; e < 8; ++e) {
        float xv = acc[tn][e] + be;
        xv = xv > 0.f ? xv : 0.f;
        upart[e] += xv * wmw;
      }
    }
    __syncthreads();
  }
  #pragma unroll
  for (int m = 1; m <= 8; m <<= 1)
    #pragma unroll
    for (int e = 0; e < 8; ++e) upart[e] += __shfl_xor(upart[e], m, 32);
  if (l15 == 0) {
    #pragma unroll
    for (int e = 0; e < 8; ++e) atomicAdd(&u_l[wm * 16 + e + hi8], upart[e]);
  }
  __syncthreads();
  if (t < 64) u[r0 + t] = u_l[t];
}

// ---- matvec + pointwise: xls = d_i*(A@(d*u))_i + d_i^2*u_i + bmu, etc. -----
__global__ void __launch_bounds__(256) matvec_kernel(const float* __restrict__ A,
                                                     const float* __restrict__ dvec,
                                                     const float* __restrict__ u,
                                                     const float* __restrict__ noise,
                                                     const float* __restrict__ bmu,
                                                     float* __restrict__ xmu_o,
                                                     float* __restrict__ xls_o,
                                                     float* __restrict__ lat_o,
                                                     float* __restrict__ lat_ws) {
  __shared__ float wl[NDIM];
  int t = threadIdx.x;
  int b = blockIdx.x >> 8;
  #pragma unroll
  for (int c = 0; c < 8; ++c) {
    int j = c * 256 + t;
    wl[j] = dvec[b * NDIM + j] * u[b * NDIM + j];
  }
  __syncthreads();
  int wv = t >> 5, lane = t & 31;
  int i = (blockIdx.x & 255) * 8 + wv;
  const float4* row = (const float4*)(A + ((size_t)b * NDIM + i) * NDIM);
  float s = 0.f;
  #pragma unroll
  for (int c = 0; c < 16; ++c) {
    float4 v = row[c * 32 + lane];
    float4 w = *(const float4*)&wl[(c * 32 + lane) * 4];
    s += v.x * w.x + v.y * w.y + v.z * w.z + v.w * w.w;
  }
  #pragma unroll
  for (int m = 16; m; m >>= 1) s += __shfl_xor(s, m, 32);
  if (lane == 0) {
    float bm  = bmu[0];
    float v   = s + wl[i];                            // +I term
    float xls = dvec[b * NDIM + i] * v + bm;
    float xm  = u[b * NDIM + i] + bm;
    float lat = xm + noise[i] * __expf(xls);
    int idx = b * NDIM + i;
    xmu_o[idx] = xm; xls_o[idx] = xls; lat_o[idx] = lat; lat_ws[idx] = lat;
  }
}

// ---- outer product + sigmoid, dual store -----------------------------------
__global__ void __launch_bounds__(256) outer_kernel(const float* __restrict__ lat,
                                                    float* __restrict__ o1,
                                                    float* __restrict__ o2) {
  int rf = blockIdx.x;
  int b = rf >> 11;
  float li = lat[rf];
  const float4* lj = (const float4*)(lat + ((size_t)b << 11));
  size_t base = (size_t)rf * NDIM;
  float4* p1 = (float4*)(o1 + base);
  float4* p2 = (float4*)(o2 + base);
  int t = threadIdx.x;
  #pragma unroll
  for (int c = 0; c < 2; ++c) {
    int q = c * 256 + t;
    float4 v = lj[q];
    float4 r;
    r.x = 1.f / (1.f + __expf(-li * v.x));
    r.y = 1.f / (1.f + __expf(-li * v.y));
    r.z = 1.f / (1.f + __expf(-li * v.z));
    r.w = 1.f / (1.f + __expf(-li * v.w));
    p1[q] = r;
    p2[q] = r;
  }
}

extern "C" void kernel_launch(void* const* d_in, const int* in_sizes, int n_in,
                              void* d_out, int out_size, void* d_ws, size_t ws_size,
                              hipStream_t stream) {
  const float* A     = (const float*)d_in[0];
  const float* h0    = (const float*)d_in[1];
  const float* Wext  = (const float*)d_in[2];
  const float* bext  = (const float*)d_in[3];
  const float* Wmu   = (const float*)d_in[4];
  const float* bmu   = (const float*)d_in[5];
  const float* noise = (const float*)d_in[6];

  float* out = (float*)d_out;
  const size_t S = (size_t)BATCH * NDIM * NDIM;       // 33,554,432
  float* o1    = out;                                 // A_hat_reshape
  float* o2    = out + S;                             // A_hat_sig
  float* xmu_o = out + 2 * S;
  float* xls_o = xmu_o + BATCH * NDIM;
  float* lat_o = xls_o + BATCH * NDIM;

  // small f32 scratch in d_ws (192 KB)
  char* ws = (char*)d_ws;
  float* dvec = (float*)(ws + 0);
  float* u    = (float*)(ws + (64 << 10));
  float* latw = (float*)(ws + (128 << 10));

  // bf16 intermediates parked inside the o2 region (fully overwritten at end)
  char* sc = (char*)o2;
  unsigned short* Wb   = (unsigned short*)(sc);                 // 1 MB
  unsigned short* h0pT = (unsigned short*)(sc + (2u << 20));    // 8 MB
  unsigned short* Hb   = (unsigned short*)(sc + (10u << 20));   // 8 MB
  unsigned short* Ab   = (unsigned short*)(sc + (18u << 20));   // 64 MB

  deg_kernel<<<2048, 256, 0, stream>>>(A, dvec, Ab);
  prep_w_kernel<<<2048, 256, 0, stream>>>(Wext, Wb, NDIM * DIN);
  prep_h0_kernel<<<256, 256, 0, stream>>>(h0, dvec, h0pT);

  dim3 g1(2, 16, 8);
  gemm1_kernel<<<g1, 256, 0, stream>>>(Ab, h0pT, h0, dvec, Hb);
  gemm2_kernel<<<256, 256, 0, stream>>>(Hb, Wb, bext, Wmu, u);
  matvec_kernel<<<2048, 256, 0, stream>>>(A, dvec, u, noise, bmu, xmu_o, xls_o, lat_o, latw);
  outer_kernel<<<BATCH * NDIM, 256, 0, stream>>>(latw, o1, o2);
}
